// crossattention_76416058130794
// MI455X (gfx1250) — compile-verified, matrix operand order checked by reading, not run on hardware
//
#include <hip/hip_runtime.h>
#include <math.h>

// ---------------------------------------------------------------------------
// Cross-attention for B=4, L=4096, D=256 (fp32 in/out) on gfx1250 (CDNA5).
// f32 -> f16 projections (WMMA f32 accum), chunked two-pass streaming softmax,
// transposed-attention output GEMM with multi-wave blocks + LDS P-tile share.
// ---------------------------------------------------------------------------

#define BATCH  4
#define SEQ    4096
#define DMODEL 256
#define SCALE  0.0625f    // 1/sqrt(256)
#define NCHUNK 4          // j-range chunks for the stats pass
#define NTOK   (BATCH * SEQ)

typedef _Float16 v16h __attribute__((ext_vector_type(16)));
typedef _Float16 v8h  __attribute__((ext_vector_type(8)));
typedef float    v8f  __attribute__((ext_vector_type(8)));
typedef float    v4f  __attribute__((ext_vector_type(4)));

static __device__ __forceinline__ v8f wmma16x16x32(v16h a, v16h b, v8f c) {
  // 8 args: (neg_a, A, neg_b, B, c_mod, C, reuse_a, reuse_b)
  return __builtin_amdgcn_wmma_f32_16x16x32_f16(false, a, false, b, (short)0, c,
                                                false, false);
}

// Load a 16x32 f16 A/B fragment where the 32-long K dimension is CONTIGUOUS in
// memory for this lane's row/column.  rowptr = &M[row][k0] for row = lane&15.
// ISA layout (16-bit A 16x32): lanes 0-15 hold K[0..7] then K[16..23] in their
// 16 halves, lanes 16-31 hold K[8..15] then K[24..31].  halfsel = lane>>4.
static __device__ __forceinline__ v16h frag_f16(const _Float16* __restrict__ rowptr,
                                                int halfsel) {
  v8h c0 = *(const v8h*)(rowptr + halfsel * 8);
  v8h c1 = *(const v8h*)(rowptr + 16 + halfsel * 8);
  v16h a;
#pragma unroll
  for (int i = 0; i < 8; ++i) { a[i] = c0[i]; a[i + 8] = c1[i]; }
  return a;
}

// Same fragment gather from an fp32 source, converting to f16 on the fly.
static __device__ __forceinline__ v16h frag_f32(const float* __restrict__ rowptr,
                                                int halfsel) {
  v4f f0 = *(const v4f*)(rowptr + halfsel * 8);
  v4f f1 = *(const v4f*)(rowptr + halfsel * 8 + 4);
  v4f f2 = *(const v4f*)(rowptr + 16 + halfsel * 8);
  v4f f3 = *(const v4f*)(rowptr + 16 + halfsel * 8 + 4);
  v16h a;
#pragma unroll
  for (int i = 0; i < 4; ++i) {
    a[i]      = (_Float16)f0[i];
    a[i + 4]  = (_Float16)f1[i];
    a[i + 8]  = (_Float16)f2[i];
    a[i + 12] = (_Float16)f3[i];
  }
  return a;
}

// ---------------------------------------------------------------------------
// Kernel 1: projection  Y = X @ W^T + b, output in f16.
//   B-operand column e of W^T is row e of W -> K-contiguous, same loader as A.
//   transposeOut=1 stores Y transposed per batch: YT[b][d][i]  (for V).
// Grid: (B*L/16, D/16), one wave per block -> one 16x16 C tile.
// ---------------------------------------------------------------------------
__global__ void __launch_bounds__(32)
proj_kernel(const float* __restrict__ X, const float* __restrict__ W,
            const float* __restrict__ bias, _Float16* __restrict__ Y,
            int transposeOut) {
  const int lane = threadIdx.x;
  const int r = lane & 15;
  const int h = lane >> 4;
  const int m0 = blockIdx.x * 16;
  const int e0 = blockIdx.y * 16;

  v8f c = {0.f, 0.f, 0.f, 0.f, 0.f, 0.f, 0.f, 0.f};
  const float* xrow = X + (size_t)(m0 + r) * DMODEL;
  const float* wrow = W + (size_t)(e0 + r) * DMODEL;
#pragma unroll
  for (int kk = 0; kk < DMODEL; kk += 32) {
    v16h a = frag_f32(xrow + kk, h);
    v16h b = frag_f32(wrow + kk, h);
    c = wmma16x16x32(a, b, c);
  }

  const float bv = bias[e0 + r];  // C layout: lane holds column n = lane&15
  if (!transposeOut) {
#pragma unroll
    for (int t = 0; t < 8; ++t) {
      const int row = m0 + t + 8 * h;
      Y[(size_t)row * DMODEL + e0 + r] = (_Float16)(c[t] + bv);
    }
  } else {
#pragma unroll
    for (int t = 0; t < 8; ++t) {
      const int row = m0 + t + 8 * h;     // global (b*L + i)
      const int b_  = row >> 12;          // / SEQ
      const int i   = row & (SEQ - 1);
      Y[(size_t)b_ * DMODEL * SEQ + (size_t)(e0 + r) * SEQ + i] =
          (_Float16)(c[t] + bv);
    }
  }
}

// ---------------------------------------------------------------------------
// Kernel 2a: partial softmax row statistics over one j-chunk.
//   m = max_j S[i,j], l = sum_j exp(S[i,j]-m) for j in this chunk.
// Streaming (m,l) kept per-lane (lane owns one j-column of the C tile),
// merged across the 16-lane group at the end.  Grid: (B*L/16, NCHUNK).
// ---------------------------------------------------------------------------
__global__ void __launch_bounds__(32)
stats_kernel(const _Float16* __restrict__ Qp, const _Float16* __restrict__ Kp,
             float* __restrict__ mPart, float* __restrict__ lPart) {
  const int lane = threadIdx.x;
  const int r = lane & 15;
  const int h = lane >> 4;
  const int m0 = blockIdx.x * 16;       // global query-row base (b*L + i0)
  const int b_ = m0 >> 12;
  const int ch = blockIdx.y;
  const int jBeg = ch * (SEQ / NCHUNK);
  const int jEnd = jBeg + (SEQ / NCHUNK);

  // Preload the 8 K-step A fragments of the fixed Q tile.
  v16h afr[8];
  const _Float16* qrow = Qp + (size_t)(m0 + r) * DMODEL;
#pragma unroll
  for (int kk = 0; kk < 8; ++kk) afr[kk] = frag_f16(qrow + kk * 32, h);

  float rm[8], rl[8];
#pragma unroll
  for (int t = 0; t < 8; ++t) { rm[t] = -3.0e38f; rl[t] = 0.f; }

  for (int j0 = jBeg; j0 < jEnd; j0 += 16) {
    const _Float16* krow = Kp + (size_t)(b_ * SEQ + j0 + r) * DMODEL;
    __builtin_prefetch(krow + 16 * DMODEL, 0, 0);  // next j-tile row
    v8f c = {0.f, 0.f, 0.f, 0.f, 0.f, 0.f, 0.f, 0.f};
#pragma unroll
    for (int kk = 0; kk < 8; ++kk) {
      v16h bf = frag_f16(krow + kk * 32, h);   // column j contiguous over d
      c = wmma16x16x32(afr[kk], bf, c);
    }
#pragma unroll
    for (int t = 0; t < 8; ++t) {
      const float v = c[t] * SCALE;
      const float nm = fmaxf(rm[t], v);
      rl[t] = rl[t] * __expf(rm[t] - nm) + __expf(v - nm);
      rm[t] = nm;
    }
  }

  // Merge (m,l) across the 16 lanes that share each row.
#pragma unroll
  for (int off = 8; off >= 1; off >>= 1) {
#pragma unroll
    for (int t = 0; t < 8; ++t) {
      const float om = __shfl_xor(rm[t], off, 16);
      const float ol = __shfl_xor(rl[t], off, 16);
      const float nm = fmaxf(rm[t], om);
      rl[t] = rl[t] * __expf(rm[t] - nm) + ol * __expf(om - nm);
      rm[t] = nm;
    }
  }
  if (r == 0) {
#pragma unroll
    for (int t = 0; t < 8; ++t) {
      const int row = m0 + t + 8 * h;
      mPart[(size_t)ch * NTOK + row] = rm[t];
      lPart[(size_t)ch * NTOK + row] = rl[t];
    }
  }
}

// ---------------------------------------------------------------------------
// Kernel 2b: merge NCHUNK partial (m,l) pairs per row (log-sum-exp merge).
// ---------------------------------------------------------------------------
__global__ void __launch_bounds__(256)
merge_kernel(const float* __restrict__ mPart, const float* __restrict__ lPart,
             float* __restrict__ mOut, float* __restrict__ lOut) {
  const int row = blockIdx.x * 256 + threadIdx.x;
  if (row >= NTOK) return;
  float m = -3.0e38f;
#pragma unroll
  for (int c = 0; c < NCHUNK; ++c) m = fmaxf(m, mPart[(size_t)c * NTOK + row]);
  float l = 0.f;
#pragma unroll
  for (int c = 0; c < NCHUNK; ++c)
    l += lPart[(size_t)c * NTOK + row] * __expf(mPart[(size_t)c * NTOK + row] - m);
  mOut[row] = m;
  lOut[row] = l;
}

// ---------------------------------------------------------------------------
// Kernel 3: out[b,j,d] = sum_i softmax(S)[i,j] * V[b,i,d]  (= attn^T @ V).
// 4 waves per block, i-step = 128.  Each wave computes T = S^T for its 32
// i-columns (probabilities land in A-row orientation and the exp(m,l)
// normalization is lane-local), all waves fill a shared 16x128 f16 P tile in
// LDS, then each wave accumulates its 4 d-tiles against K-contiguous VpT.
// Per wave: 16 T-WMMAs + 16 out-WMMAs per i-step; 32 acc VGPRs.
// Grid: (L/16 j-tiles, B) x 128 threads.
// ---------------------------------------------------------------------------
__global__ void __launch_bounds__(128)
out_kernel(const _Float16* __restrict__ Qp, const _Float16* __restrict__ Kp,
           const _Float16* __restrict__ VpT, const float* __restrict__ mIn,
           const float* __restrict__ lIn, float* __restrict__ out) {
  __shared__ __align__(16) _Float16 pLds[16 * 128];   // P tile: 16 j x 128 i

  const int lane = threadIdx.x & 31;
  const int wave = threadIdx.x >> 5;                  // 0..3
  const int r = lane & 15;
  const int h = lane >> 4;
  const int j0 = blockIdx.x * 16;
  const int b_ = blockIdx.y;
  const size_t rowbase = (size_t)b_ * SEQ;

  // Fixed A operand of the T-GEMM: K rows j0..j0+15 (per-wave copy).
  v16h kfr[8];
  const _Float16* krow = Kp + (rowbase + j0 + r) * DMODEL;
#pragma unroll
  for (int kk = 0; kk < 8; ++kk) kfr[kk] = frag_f16(krow + kk * 32, h);

  v8f acc[4];
#pragma unroll
  for (int dl = 0; dl < 4; ++dl)
#pragma unroll
    for (int t = 0; t < 8; ++t) acc[dl][t] = 0.f;

  const _Float16* vbase = VpT + (size_t)b_ * DMODEL * SEQ;

  for (int i0 = 0; i0 < SEQ; i0 += 128) {
    const int ic = i0 + wave * 32;                    // this wave's 32 columns

    // --- T = S^T tiles for i-columns [ic, ic+16) and [ic+16, ic+32) ---
    v8f c0 = {0.f, 0.f, 0.f, 0.f, 0.f, 0.f, 0.f, 0.f};
    v8f c1 = {0.f, 0.f, 0.f, 0.f, 0.f, 0.f, 0.f, 0.f};
    const _Float16* q0 = Qp + (rowbase + ic + r) * DMODEL;
    const _Float16* q1 = Qp + (rowbase + ic + 16 + r) * DMODEL;
    __builtin_prefetch(q0 + 128 * DMODEL, 0, 0);      // next i-step rows
#pragma unroll
    for (int kk = 0; kk < 8; ++kk) c0 = wmma16x16x32(kfr[kk], frag_f16(q0 + kk * 32, h), c0);
#pragma unroll
    for (int kk = 0; kk < 8; ++kk) c1 = wmma16x16x32(kfr[kk], frag_f16(q1 + kk * 32, h), c1);

    // Per-lane softmax normalization (lane owns i-column ic + r [+16]).
    const float mm0 = mIn[rowbase + ic + r];
    const float il0 = 1.0f / lIn[rowbase + ic + r];
    const float mm1 = mIn[rowbase + ic + 16 + r];
    const float il1 = 1.0f / lIn[rowbase + ic + 16 + r];

    __syncthreads();  // previous iteration's LDS reads complete
#pragma unroll
    for (int t = 0; t < 8; ++t) {
      const int jr = (t + 8 * h) * 128 + wave * 32;
      pLds[jr + r]      = (_Float16)(__expf(c0[t] * SCALE - mm0) * il0);
      pLds[jr + r + 16] = (_Float16)(__expf(c1[t] * SCALE - mm1) * il1);
    }
    __syncthreads();

    // acc[dl] += P(16x128) @ V(128x16) for this wave's 4 d-tiles.
#pragma unroll
    for (int ks = 0; ks < 4; ++ks) {
      v16h ap = frag_f16(pLds + r * 128 + ks * 32, h);
#pragma unroll
      for (int dl = 0; dl < 4; ++dl) {
        const int d0 = (wave * 4 + dl) * 16;
        v16h bv = frag_f16(vbase + (size_t)(d0 + r) * SEQ + i0 + ks * 32, h);
        acc[dl] = wmma16x16x32(ap, bv, acc[dl]);
      }
    }
  }

  // Store fp32 result: out[b][j][d].
#pragma unroll
  for (int dl = 0; dl < 4; ++dl)
#pragma unroll
    for (int t = 0; t < 8; ++t) {
      const int j = j0 + t + 8 * h;
      const int d = (wave * 4 + dl) * 16 + r;
      out[((size_t)b_ * SEQ + j) * DMODEL + d] = acc[dl][t];
    }
}

// ---------------------------------------------------------------------------
extern "C" void kernel_launch(void* const* d_in, const int* in_sizes, int n_in,
                              void* d_out, int out_size, void* d_ws, size_t ws_size,
                              hipStream_t stream) {
  const float* q  = (const float*)d_in[0];
  const float* k  = (const float*)d_in[1];
  const float* v  = (const float*)d_in[2];
  const float* Wq = (const float*)d_in[3];
  const float* bq = (const float*)d_in[4];
  const float* Wk = (const float*)d_in[5];
  const float* bk = (const float*)d_in[6];
  const float* Wv = (const float*)d_in[7];
  const float* bv = (const float*)d_in[8];

  _Float16* Qp  = (_Float16*)d_ws;
  _Float16* Kp  = Qp + (size_t)NTOK * DMODEL;
  _Float16* VpT = Kp + (size_t)NTOK * DMODEL;          // stored as [b][d][i]
  float* mPart  = (float*)(VpT + (size_t)NTOK * DMODEL);
  float* lPart  = mPart + (size_t)NCHUNK * NTOK;
  float* mArr   = lPart + (size_t)NCHUNK * NTOK;
  float* lArr   = mArr + NTOK;

  const dim3 gproj(NTOK / 16, DMODEL / 16);
  proj_kernel<<<gproj, 32, 0, stream>>>(q, Wq, bq, Qp, 0);
  proj_kernel<<<gproj, 32, 0, stream>>>(k, Wk, bk, Kp, 0);
  proj_kernel<<<gproj, 32, 0, stream>>>(v, Wv, bv, VpT, 1);

  stats_kernel<<<dim3(NTOK / 16, NCHUNK), 32, 0, stream>>>(Qp, Kp, mPart, lPart);
  merge_kernel<<<dim3((NTOK + 255) / 256), 256, 0, stream>>>(mPart, lPart, mArr, lArr);

  out_kernel<<<dim3(SEQ / 16, BATCH), 128, 0, stream>>>(Qp, Kp, VpT, mArr, lArr,
                                                        (float*)d_out);
}
